// SparseNeighborhoodModel_20444044329128
// MI455X (gfx1250) — matrix operand order repaired and checked by reading, not compiled, not run on hardware
//
#include <hip/hip_runtime.h>
#include <hip/hip_bf16.h>

// ---------------------------------------------------------------------------
// SparseNeighborhoodModel on gfx1250 (MI455X).
// Dense GEMMs run on v_wmma_f32_16x16x32_bf16 (wave32, 16x16 tiles, f32 acc).
// A strips staged in LDS via GLOBAL_LOAD_ASYNC_TO_LDS_B128 (ASYNCcnt) when
// the builtin is available. Graph ops are bandwidth-bound kernels designed
// around the 192MB L2.
// ---------------------------------------------------------------------------

typedef __attribute__((ext_vector_type(16))) __bf16 bf16x16;
typedef __attribute__((ext_vector_type(8)))  float  floatx8;
typedef __attribute__((ext_vector_type(4)))  int    v4i_t;

#define DH 256
#define RED_NB 512

#if defined(__HIP_DEVICE_COMPILE__) && __has_builtin(__builtin_amdgcn_global_load_async_to_lds_b128)
#define USE_ASYNC_LDS 1
#endif

__device__ __forceinline__ unsigned short f32_bf16(float f) {
  union { float f; unsigned u; } a; a.f = f;
  unsigned u = a.u;
  unsigned round = 0x7FFFu + ((u >> 16) & 1u);   // round-to-nearest-even
  return (unsigned short)((u + round) >> 16);
}

__device__ __forceinline__ void async_wait_zero() {
#if defined(USE_ASYNC_LDS)
#if __has_builtin(__builtin_amdgcn_s_wait_asynccnt)
  __builtin_amdgcn_s_wait_asynccnt(0);
#else
  asm volatile("s_wait_asynccnt 0x0" ::: "memory");
#endif
#endif
}

// ---------------------------------------------------------------------------
// WMMA GEMM:  C(M x 256) = act( A(M x Kpad, bf16) @ Wp + bias [+ Cin] )
// Wp is pre-packed in the per-lane B-fragment layout (see pack_weight_bf16).
// Block: 256 threads = 8 waves; wave w owns column tiles 2w, 2w+1 and BOTH
// 16-row tiles of a 32-row strip (B fragments reused across the two row
// tiles -> 4 wmma per K-step per wave).
// Grid.x: ceil(M/32) row strips. A strip staged in LDS (row stride Kpad+8
// to spread rows across banks for the ds_load_b128 fragment reads).
// ---------------------------------------------------------------------------
__device__ __forceinline__ void gemm_store_tile(
    floatx8 acc, int growbase, int col, int M, int hi,
    const float* __restrict__ bias, const float* __restrict__ Cin,
    float* __restrict__ Cout, int act)
{
  float b = bias ? bias[col] : 0.f;
#pragma unroll
  for (int i = 0; i < 8; ++i) {
    int grow = growbase + i + 8 * hi;
    if (grow >= M) continue;
    float v = acc[i] + b;
    if (Cin)  v += Cin[(size_t)grow * DH + col];
    if (act == 1)      v = fmaxf(v, 0.f);
    else if (act == 2) v = tanhf(v);
    else if (act == 3) v = 1.f / (1.f + __expf(-v));
    Cout[(size_t)grow * DH + col] = v;
  }
}

__global__ __launch_bounds__(256) void gemm_wmma_bf16(
    const unsigned short* __restrict__ Abf, int M, int Kpad,
    const unsigned short* __restrict__ Wp,
    const float* __restrict__ bias,
    const float* __restrict__ Cin,
    float* __restrict__ Cout,
    int act)   // 0 none, 1 relu, 2 tanh, 3 sigmoid
{
  __shared__ __align__(16) unsigned short ldsA[32 * 264];
  const int stride  = Kpad + 8;
  const int rowbase = blockIdx.x * 32;
  const int tid     = threadIdx.x;

  // Stage the 32 x Kpad bf16 A strip into LDS (32B chunks).
  const int chunksPerRow = Kpad >> 4;
  const int nchunk = 32 * chunksPerRow;
  for (int c = tid; c < nchunk; c += 256) {
    int row = c / chunksPerRow;
    int ch  = c % chunksPerRow;
    unsigned short* l = &ldsA[row * stride + ch * 16];
    if (rowbase + row < M) {
      const unsigned short* g = Abf + (size_t)(rowbase + row) * Kpad + ch * 16;
#if defined(USE_ASYNC_LDS)
      // ASYNCcnt-tracked direct copy to LDS, no VGPR round trip.
      __attribute__((address_space(1))) v4i_t* gv =
          (__attribute__((address_space(1))) v4i_t*)(v4i_t*)(unsigned short*)g;
      __attribute__((address_space(3))) v4i_t* lv =
          (__attribute__((address_space(3))) v4i_t*)(v4i_t*)l;
      __builtin_amdgcn_global_load_async_to_lds_b128(gv, lv, 0, 0);
      __builtin_amdgcn_global_load_async_to_lds_b128(gv, lv, 16, 0);
#else
      const uint4* src = reinterpret_cast<const uint4*>(g);
      uint4* dst = reinterpret_cast<uint4*>(l);
      dst[0] = src[0]; dst[1] = src[1];
#endif
    } else {
      uint4* dst = reinterpret_cast<uint4*>(l);
      uint4 zz = make_uint4(0, 0, 0, 0);
      dst[0] = zz; dst[1] = zz;
    }
  }
  async_wait_zero();
  __syncthreads();

  const int wave = tid >> 5;
  const int lane = tid & 31;
  const int lm   = lane & 15;
  const int hi   = lane >> 4;
  const int ksteps = Kpad >> 5;
  const int ct0 = wave * 2, ct1 = wave * 2 + 1;

  floatx8 z8 = {0.f, 0.f, 0.f, 0.f, 0.f, 0.f, 0.f, 0.f};
  floatx8 acc00 = z8, acc01 = z8, acc10 = z8, acc11 = z8;

  for (int ks = 0; ks < ksteps; ++ks) {
    // A fragments for both row tiles.  16-bit 16x32 layout: lane half hi:
    // j<8 -> K=ks*32+8*hi+j, j>=8 -> K=ks*32+16+8*hi+(j-8).
    union { uint4 u[2]; bf16x16 v; } a0, a1, b0, b1;
    const unsigned short* ap0 = &ldsA[lm * stride + ks * 32 + 8 * hi];
    a0.u[0] = *reinterpret_cast<const uint4*>(ap0);
    a0.u[1] = *reinterpret_cast<const uint4*>(ap0 + 16);
    const unsigned short* ap1 = &ldsA[(16 + lm) * stride + ks * 32 + 8 * hi];
    a1.u[0] = *reinterpret_cast<const uint4*>(ap1);
    a1.u[1] = *reinterpret_cast<const uint4*>(ap1 + 16);
    // B fragments: packed contiguous 32B per lane, shared by both row tiles.
    const uint4* bp0 = reinterpret_cast<const uint4*>(
        Wp + (((size_t)ct0 * ksteps + ks) * 32 + lane) * 16);
    b0.u[0] = bp0[0]; b0.u[1] = bp0[1];
    const uint4* bp1 = reinterpret_cast<const uint4*>(
        Wp + (((size_t)ct1 * ksteps + ks) * 32 + lane) * 16);
    b1.u[0] = bp1[0]; b1.u[1] = bp1[1];

    acc00 = __builtin_amdgcn_wmma_f32_16x16x32_bf16(
        false, a0.v, false, b0.v, (short)0, acc00, false, false);
    acc01 = __builtin_amdgcn_wmma_f32_16x16x32_bf16(
        false, a0.v, false, b1.v, (short)0, acc01, false, false);
    acc10 = __builtin_amdgcn_wmma_f32_16x16x32_bf16(
        false, a1.v, false, b0.v, (short)0, acc10, false, false);
    acc11 = __builtin_amdgcn_wmma_f32_16x16x32_bf16(
        false, a1.v, false, b1.v, (short)0, acc11, false, false);
  }

  // C/D layout: VGPR i, lane half hi -> row = i + 8*hi, col = ct*16 + (lane&15)
  gemm_store_tile(acc00, rowbase,      ct0 * 16 + lm, M, hi, bias, Cin, Cout, act);
  gemm_store_tile(acc01, rowbase,      ct1 * 16 + lm, M, hi, bias, Cin, Cout, act);
  gemm_store_tile(acc10, rowbase + 16, ct0 * 16 + lm, M, hi, bias, Cin, Cout, act);
  gemm_store_tile(acc11, rowbase + 16, ct1 * 16 + lm, M, hi, bias, Cin, Cout, act);
}

// Pack W (K x 256 f32, row-major) into the per-lane B-fragment bf16 layout:
// Wp[((ct*ksteps + ks)*32 + lane)*16 + j] = W[ks*32 + 8*(lane>>4) + (j&7) + 16*(j>>3)][ct*16 + (lane&15)]
__global__ void pack_weight_bf16(const float* __restrict__ W, int K, int ksteps,
                                 unsigned short* __restrict__ Wp) {
  int idx = blockIdx.x * blockDim.x + threadIdx.x;
  int total = 16 * ksteps * 512;
  if (idx >= total) return;
  int j    = idx & 15;
  int lane = (idx >> 4) & 31;
  int rest = idx >> 9;
  int ks = rest % ksteps;
  int ct = rest / ksteps;
  int n  = ct * 16 + (lane & 15);
  int k  = ks * 32 + 8 * (lane >> 4) + (j & 7) + 16 * (j >> 3);
  float v = (k < K) ? W[(size_t)k * DH + n] : 0.f;
  Wp[idx] = f32_bf16(v);
}

// ------------------------- elementwise / conversion ------------------------
__global__ void cvt_f32_bf16_k(const float* __restrict__ a,
                               unsigned short* __restrict__ o, int n) {
  int i = blockIdx.x * blockDim.x + threadIdx.x;
  if (i < n) o[i] = f32_bf16(a[i]);
}
__global__ void cvt_mul_bf16_k(const float* __restrict__ a, const float* __restrict__ b,
                               unsigned short* __restrict__ o, int n) {
  int i = blockIdx.x * blockDim.x + threadIdx.x;
  if (i < n) o[i] = f32_bf16(a[i] * b[i]);
}
// [emb(32) | feat(8) | zeros(24)] per node, bf16 (K padded 40 -> 64)
__global__ void build_enc_k(const float* __restrict__ emb, const float* __restrict__ feat,
                            unsigned short* __restrict__ o, int N) {
  int i = blockIdx.x * blockDim.x + threadIdx.x;
  if (i >= N * 64) return;
  int n = i >> 6, c = i & 63;
  float v = (c < 32) ? emb[n * 32 + c] : ((c < 40) ? feat[n * 8 + c - 32] : 0.f);
  o[i] = f32_bf16(v);
}
__global__ void gru_update_k(float* __restrict__ h, const float* __restrict__ z,
                             const float* __restrict__ cand, int n) {
  int i = blockIdx.x * blockDim.x + threadIdx.x;
  if (i < n) { float zz = z[i]; h[i] = zz * h[i] + (1.f - zz) * cand[i]; }
}
__global__ void zero_u32_k(unsigned* __restrict__ p, int n) {
  int i = blockIdx.x * blockDim.x + threadIdx.x;
  if (i < n) p[i] = 0u;
}

// ----------------------------- graph structure -----------------------------
__global__ void count_deg_k(const int* __restrict__ src, int* __restrict__ counts, int E) {
  int i = blockIdx.x * blockDim.x + threadIdx.x;
  if (i < E) atomicAdd(&counts[src[i]], 1);
}
__global__ __launch_bounds__(256) void scan_offsets_k(const int* __restrict__ counts,
                                                      int* __restrict__ off, int N) {
  __shared__ int sdata[256];
  __shared__ int carry;
  if (threadIdx.x == 0) carry = 0;
  __syncthreads();
  for (int base = 0; base < N; base += 256) {
    int i = base + threadIdx.x;
    sdata[threadIdx.x] = (i < N) ? counts[i] : 0;
    __syncthreads();
    for (int o = 1; o < 256; o <<= 1) {
      int t = (threadIdx.x >= o) ? sdata[threadIdx.x - o] : 0;
      __syncthreads();
      sdata[threadIdx.x] += t;
      __syncthreads();
    }
    if (i < N) off[i + 1] = carry + sdata[threadIdx.x];
    __syncthreads();
    if (threadIdx.x == 255) carry += sdata[255];
    __syncthreads();
  }
  if (threadIdx.x == 0) off[0] = 0;
}

// agg[n] = sum over e in [off[n],off[n+1]) of t[dst[e]]   (deterministic CSR)
__global__ __launch_bounds__(256) void aggregate_k(const float* __restrict__ t,
                                                   const int* __restrict__ off,
                                                   const int* __restrict__ dst,
                                                   float* __restrict__ agg, int N) {
  int n = blockIdx.x;
  int c = threadIdx.x;
  int s = off[n], e = off[n + 1];
  float acc = 0.f;
  for (int i = s; i < e; ++i)
    acc += t[(size_t)dst[i] * DH + c];
  agg[(size_t)n * DH + c] = acc;
}

// --------------------------- decode / softmax ------------------------------
__global__ void transpose_dec_k(const float* __restrict__ Wdec /*256 x N*/,
                                float* __restrict__ WdT /*N x 256*/, int N) {
  int i = blockIdx.x * blockDim.x + threadIdx.x;
  if (i >= N * DH) return;
  int n = i / DH, k = i % DH;
  WdT[i] = Wdec[(size_t)k * N + n];
}
// logits[e] = dot(h[src], WdT[dst]) + b_dec[dst]   (one wave32 per edge)
__global__ __launch_bounds__(256) void edge_logits_k(
    const float* __restrict__ h, const float* __restrict__ WdT,
    const float* __restrict__ bdec, const int* __restrict__ src,
    const int* __restrict__ dst, float* __restrict__ out, int E) {
  int e = blockIdx.x * 8 + (threadIdx.x >> 5);
  if (e >= E) return;
  int lane = threadIdx.x & 31;
  const float* hr = h + (size_t)src[e] * DH;
  const float* wr = WdT + (size_t)dst[e] * DH;
  float s = 0.f;
  for (int k = lane; k < DH; k += 32) s += hr[k] * wr[k];
  for (int o = 16; o > 0; o >>= 1) s += __shfl_xor(s, o, 32);
  if (lane == 0) out[e] = s + bdec[dst[e]];
}
__global__ __launch_bounds__(256) void seg_stats_k(const float* __restrict__ logits,
                                                   const int* __restrict__ off,
                                                   float* __restrict__ segmax,
                                                   float* __restrict__ segsum, int N) {
  int n = blockIdx.x * 8 + (threadIdx.x >> 5);
  if (n >= N) return;
  int lane = threadIdx.x & 31;
  int s = off[n], t = off[n + 1];
  float mx = -3.4e38f;
  for (int i = s + lane; i < t; i += 32) mx = fmaxf(mx, logits[i]);
  for (int o = 16; o > 0; o >>= 1) mx = fmaxf(mx, __shfl_xor(mx, o, 32));
  float sum = 0.f;
  for (int i = s + lane; i < t; i += 32) sum += __expf(logits[i] - mx);
  for (int o = 16; o > 0; o >>= 1) sum += __shfl_xor(sum, o, 32);
  if (lane == 0) { segmax[n] = mx; segsum[n] = sum; }
}
__global__ void softmax_w_k(const float* __restrict__ logits, const int* __restrict__ src,
                            const float* __restrict__ segmax, const float* __restrict__ segsum,
                            float* __restrict__ w, int E) {
  int i = blockIdx.x * blockDim.x + threadIdx.x;
  if (i < E) { int s = src[i]; w[i] = __expf(logits[i] - segmax[s]) / segsum[s]; }
}

// ------------------------------ flow iteration -----------------------------
__global__ void scatter_flow_k(const float* __restrict__ flow, const int* __restrict__ dst,
                               float* __restrict__ inflow, int E) {
  int i = blockIdx.x * blockDim.x + threadIdx.x;
  if (i < E) atomicAdd(&inflow[dst[i]], flow[i]);
}
__global__ void update_flow_k(const float* __restrict__ w, const float* __restrict__ inflow,
                              const float* __restrict__ dem, const int* __restrict__ src,
                              float* __restrict__ flow, int E) {
  int i = blockIdx.x * blockDim.x + threadIdx.x;
  if (i < E) { int s = src[i]; flow[i] = w[i] * fmaxf(inflow[s] - dem[s], 0.f); }
}

// ------------------------------- dual head ---------------------------------
__global__ __launch_bounds__(256) void dual_head_k(const float* __restrict__ q,
                                                   const float* __restrict__ wd2,
                                                   const float* __restrict__ bd2,
                                                   float* __restrict__ dv, int N) {
  int n = blockIdx.x * 8 + (threadIdx.x >> 5);
  if (n >= N) return;
  int lane = threadIdx.x & 31;
  const float* qr = q + (size_t)n * DH;
  float s = 0.f;
  for (int k = lane; k < DH; k += 32) s += qr[k] * wd2[k];
  for (int o = 16; o > 0; o >>= 1) s += __shfl_xor(s, o, 32);
  if (lane == 0) dv[n] = s + bd2[0];
}

// --------------------- deterministic two-stage reductions ------------------
__device__ __forceinline__ void block_reduce_store(float a, float* part) {
  __shared__ float s[256];
  s[threadIdx.x] = a; __syncthreads();
  for (int o = 128; o > 0; o >>= 1) {
    if (threadIdx.x < (unsigned)o) s[threadIdx.x] += s[threadIdx.x + o];
    __syncthreads();
  }
  if (threadIdx.x == 0) part[blockIdx.x] = s[0];
}
__global__ __launch_bounds__(256) void red_flowsq_k(const float* __restrict__ flow, int E,
                                                    float* __restrict__ part) {
  float a = 0.f;
  for (int i = blockIdx.x * 256 + threadIdx.x; i < E; i += gridDim.x * 256) {
    float f = flow[i]; a += f * f;
  }
  block_reduce_store(a, part);
}
__global__ __launch_bounds__(256) void red_dualedge_k(const float* __restrict__ dv,
                                                      const int* __restrict__ src,
                                                      const int* __restrict__ dst, int E,
                                                      float* __restrict__ part) {
  float a = 0.f;
  for (int i = blockIdx.x * 256 + threadIdx.x; i < E; i += gridDim.x * 256) {
    float dd = dv[src[i]] - dv[dst[i]];
    float df = fmaxf(dd * 0.5f, 0.f);        // COST_C = 1
    a += df * df - df * dd;
  }
  block_reduce_store(a, part);
}
__global__ __launch_bounds__(256) void red_nodedvd_k(const float* __restrict__ dv,
                                                     const float* __restrict__ dem, int N,
                                                     float* __restrict__ part) {
  float a = 0.f;
  for (int i = blockIdx.x * 256 + threadIdx.x; i < N; i += gridDim.x * 256)
    a += dv[i] * dem[i];
  block_reduce_store(a, part);
}
// loss = flow_cost - (edge_dual - node_dual) = P0 - P1 + P2
__global__ __launch_bounds__(256) void final_loss_k(const float* __restrict__ p0,
                                                    const float* __restrict__ p1,
                                                    const float* __restrict__ p2, int nb,
                                                    float* __restrict__ out) {
  __shared__ float s[256];
  float a = 0.f;
  for (int i = threadIdx.x; i < nb; i += 256) a += p0[i] - p1[i] + p2[i];
  s[threadIdx.x] = a; __syncthreads();
  for (int o = 128; o > 0; o >>= 1) {
    if (threadIdx.x < (unsigned)o) s[threadIdx.x] += s[threadIdx.x + o];
    __syncthreads();
  }
  if (threadIdx.x == 0) out[0] = s[0];
}
__global__ void copy_flow_k(const float* __restrict__ flow, float* __restrict__ out, int E) {
  int i = blockIdx.x * blockDim.x + threadIdx.x;
  if (i < E) out[1 + i] = flow[i];
}

// ===========================================================================
extern "C" void kernel_launch(void* const* d_in, const int* in_sizes, int n_in,
                              void* d_out, int out_size, void* d_ws, size_t ws_size,
                              hipStream_t stream) {
  (void)n_in; (void)out_size; (void)ws_size;
  const float* feat  = (const float*)d_in[0];
  const float* emb   = (const float*)d_in[1];
  const float* dem   = (const float*)d_in[2];
  const int*   esrc  = (const int*)d_in[3];
  const int*   edst  = (const int*)d_in[4];
  const float* W_enc = (const float*)d_in[5];
  const float* b_enc = (const float*)d_in[6];
  const float* W_agg_in  = (const float*)d_in[7];
  const float* W_agg_out = (const float*)d_in[8];
  const float* b_agg = (const float*)d_in[9];
  const float* W_z = (const float*)d_in[10];
  const float* U_z = (const float*)d_in[11];
  const float* b_z = (const float*)d_in[12];
  const float* W_r = (const float*)d_in[13];
  const float* U_r = (const float*)d_in[14];
  const float* b_r = (const float*)d_in[15];
  const float* W_h = (const float*)d_in[16];
  const float* U_h = (const float*)d_in[17];
  const float* b_h = (const float*)d_in[18];
  const float* W_dec = (const float*)d_in[19];
  const float* b_dec = (const float*)d_in[20];
  const float* W_d1  = (const float*)d_in[21];
  const float* b_d1  = (const float*)d_in[22];
  const float* W_d2  = (const float*)d_in[23];
  const float* b_d2  = (const float*)d_in[24];

  const int N = in_sizes[2];        // demands is (N,1)
  const int E = in_sizes[3];
  const size_t ND = (size_t)N * DH;
  float* out = (float*)d_out;

  // -------- workspace bump allocator --------
  char* base = (char*)d_ws;
  size_t woff = 0;
  auto alloc = [&](size_t bytes) -> void* {
    void* r = base + woff; woff += (bytes + 255) & ~(size_t)255; return r;
  };
  float* h    = (float*)alloc(ND * 4);
  float* t    = (float*)alloc(ND * 4);    // also: cand buffer, dual q buffer
  float* agg  = (float*)alloc(ND * 4);    // also: W_dec^T after GNN layers
  float* nxt  = (float*)alloc(ND * 4);
  float* gz   = (float*)alloc(ND * 4);
  float* gr   = (float*)alloc(ND * 4);
  unsigned short* hbf   = (unsigned short*)alloc(ND * 2);
  unsigned short* nxtbf = (unsigned short*)alloc(ND * 2);
  unsigned short* xbf   = (unsigned short*)alloc(ND * 2);
  unsigned short* encbf = (unsigned short*)alloc((size_t)N * 64 * 2);
  const size_t WP256 = 16 * 8 * 512;      // packed u16 elements for K=256
  unsigned short* WpEnc    = (unsigned short*)alloc(16 * 2 * 512 * 2);
  unsigned short* WpAggIn  = (unsigned short*)alloc(WP256 * 2);
  unsigned short* WpAggOut = (unsigned short*)alloc(WP256 * 2);
  unsigned short* WpZ  = (unsigned short*)alloc(WP256 * 2);
  unsigned short* WpUz = (unsigned short*)alloc(WP256 * 2);
  unsigned short* WpR  = (unsigned short*)alloc(WP256 * 2);
  unsigned short* WpUr = (unsigned short*)alloc(WP256 * 2);
  unsigned short* WpH  = (unsigned short*)alloc(WP256 * 2);
  unsigned short* WpUh = (unsigned short*)alloc(WP256 * 2);
  unsigned short* WpD1 = (unsigned short*)alloc(WP256 * 2);
  float* eLog   = (float*)alloc((size_t)E * 4);
  float* eW     = (float*)alloc((size_t)E * 4);
  float* flow   = (float*)alloc((size_t)E * 4);
  float* inflow = (float*)alloc((size_t)N * 4);
  float* segmax = (float*)alloc((size_t)N * 4);
  float* segsum = (float*)alloc((size_t)N * 4);
  int*   counts = (int*)alloc((size_t)N * 4);
  int*   offs   = (int*)alloc((size_t)(N + 1) * 4);
  float* dv     = (float*)alloc((size_t)N * 4);
  float* part0  = (float*)alloc(RED_NB * 4);
  float* part1  = (float*)alloc(RED_NB * 4);
  float* part2  = (float*)alloc(RED_NB * 4);
  float* WdT = agg;   // alias (free after GNN layers)
  float* q   = t;     // alias (free after GNN layers)

  const int EB = (E + 255) / 256;
  const int NDB = (int)((ND + 255) / 256);
  auto gemm = [&](const unsigned short* A, int Kpad, const unsigned short* Wp,
                  const float* bias, const float* Cin, float* Cout, int act) {
    gemm_wmma_bf16<<<dim3((N + 31) / 32), 256, 0, stream>>>(A, N, Kpad, Wp, bias, Cin, Cout, act);
  };
  auto pack = [&](const float* W, int K, int ksteps, unsigned short* Wp) {
    int total = 16 * ksteps * 512;
    pack_weight_bf16<<<(total + 255) / 256, 256, 0, stream>>>(W, K, ksteps, Wp);
  };

  // -------- pack all weights to bf16 B-fragment layout --------
  pack(W_enc, 40, 2, WpEnc);
  pack(W_agg_in, 256, 8, WpAggIn);  pack(W_agg_out, 256, 8, WpAggOut);
  pack(W_z, 256, 8, WpZ);           pack(U_z, 256, 8, WpUz);
  pack(W_r, 256, 8, WpR);           pack(U_r, 256, 8, WpUr);
  pack(W_h, 256, 8, WpH);           pack(U_h, 256, 8, WpUh);
  pack(W_d1, 256, 8, WpD1);

  // -------- CSR offsets from sorted edge_src --------
  zero_u32_k<<<(N + 255) / 256, 256, 0, stream>>>((unsigned*)counts, N);
  count_deg_k<<<EB, 256, 0, stream>>>(esrc, counts, E);
  scan_offsets_k<<<1, 256, 0, stream>>>(counts, offs, N);

  // -------- encoder: h = relu([emb|feat] @ W_enc + b_enc) --------
  build_enc_k<<<(N * 64 + 255) / 256, 256, 0, stream>>>(emb, feat, encbf, N);
  gemm(encbf, 64, WpEnc, b_enc, nullptr, h, 1);

  // -------- 3 GNN layers with GRU update --------
  for (int layer = 0; layer < 3; ++layer) {
    cvt_f32_bf16_k<<<NDB, 256, 0, stream>>>(h, hbf, (int)ND);
    gemm(hbf, 256, WpAggIn, nullptr, nullptr, t, 0);                 // t = h @ W_agg_in
    aggregate_k<<<N, 256, 0, stream>>>(t, offs, edst, agg, N);       // CSR segment-sum
    cvt_f32_bf16_k<<<NDB, 256, 0, stream>>>(agg, xbf, (int)ND);
    gemm(xbf, 256, WpAggOut, b_agg, nullptr, nxt, 2);                // nxt = tanh(...)
    cvt_f32_bf16_k<<<NDB, 256, 0, stream>>>(nxt, nxtbf, (int)ND);
    gemm(nxtbf, 256, WpZ, nullptr, nullptr, gz, 0);
    gemm(hbf, 256, WpUz, b_z, gz, gz, 3);                            // z = sigmoid(...)
    gemm(nxtbf, 256, WpR, nullptr, nullptr, gr, 0);
    gemm(hbf, 256, WpUr, b_r, gr, gr, 3);                            // r = sigmoid(...)
    cvt_mul_bf16_k<<<NDB, 256, 0, stream>>>(gr, h, xbf, (int)ND);    // bf16(r*h)
    gemm(nxtbf, 256, WpH, nullptr, nullptr, t, 0);
    gemm(xbf, 256, WpUh, b_h, t, t, 2);                              // cand = tanh(...)
    gru_update_k<<<NDB, 256, 0, stream>>>(h, gz, t, (int)ND);        // h = z*h + (1-z)*cand
  }

  // -------- edge logits + segment softmax (avoids the N x N decode GEMM) ----
  transpose_dec_k<<<(int)((ND + 255) / 256), 256, 0, stream>>>(W_dec, WdT, N);
  edge_logits_k<<<(E + 7) / 8, 256, 0, stream>>>(h, WdT, b_dec, esrc, edst, eLog, E);
  seg_stats_k<<<(N + 7) / 8, 256, 0, stream>>>(eLog, offs, segmax, segsum, N);
  softmax_w_k<<<EB, 256, 0, stream>>>(eLog, esrc, segmax, segsum, eW, E);

  // -------- flow iterations --------
  zero_u32_k<<<EB, 256, 0, stream>>>((unsigned*)flow, E);
  for (int it = 0; it < 10; ++it) {
    zero_u32_k<<<(N + 255) / 256, 256, 0, stream>>>((unsigned*)inflow, N);
    scatter_flow_k<<<EB, 256, 0, stream>>>(flow, edst, inflow, E);
    update_flow_k<<<EB, 256, 0, stream>>>(eW, inflow, dem, esrc, flow, E);
  }

  // -------- dual head --------
  cvt_f32_bf16_k<<<NDB, 256, 0, stream>>>(h, hbf, (int)ND);
  gemm(hbf, 256, WpD1, b_d1, nullptr, q, 1);                         // q = relu(h@W_d1+b_d1)
  dual_head_k<<<(N + 7) / 8, 256, 0, stream>>>(q, W_d2, b_d2, dv, N);

  // -------- loss and output --------
  red_flowsq_k<<<RED_NB, 256, 0, stream>>>(flow, E, part0);
  red_dualedge_k<<<RED_NB, 256, 0, stream>>>(dv, esrc, edst, E, part1);
  red_nodedvd_k<<<RED_NB, 256, 0, stream>>>(dv, dem, N, part2);
  final_loss_k<<<1, 256, 0, stream>>>(part0, part1, part2, RED_NB, out);
  copy_flow_k<<<EB, 256, 0, stream>>>(flow, out, E);
}
